// SumThenReg2RpeSelfAttention_7103875908188
// MI455X (gfx1250) — compile-verified, matrix operand order checked by reading, not run on hardware
//
#include <hip/hip_runtime.h>
#include <hip/hip_bf16.h>

typedef _Float16 f16;
typedef __attribute__((ext_vector_type(16))) _Float16 v16h;
typedef __attribute__((ext_vector_type(8)))  _Float16 v8h;
typedef __attribute__((ext_vector_type(8)))  float    v8f;

#define NH 16       // heads
#define HD 64       // head dim
#define ED 1024     // embed dim
#define RL 2048     // reg len
#define SL 64       // sum len
#define BS 2        // batch
#define ZN 64       // num rel emb
#define QK_SCALE 0.125f
#define LN_EPS 1e-5f

// ---------------------------------------------------------------------------
// WMMA helpers (wave32).  16x16x32 f16 -> f32.
// A fragment layout per CDNA5 ISA 7.12.2 (16-bit A 16x32): lane m=lane&15,
// half hh=lane>>4 holds K = {8*hh .. 8*hh+7} and {16+8*hh .. 16+8*hh+7}
// -> two contiguous 8-half runs => two 16B LDS loads.
// B operand tiles are always STORED [n][k] in LDS, so the same loader serves
// as the B-fragment loader (symmetric lane layout).
// C/D layout: vgpr j, lane: m = j + 8*(lane>>4), n = lane&15.
// ---------------------------------------------------------------------------
__device__ __forceinline__ v8f zero_v8f() {
  v8f z;
#pragma unroll
  for (int i = 0; i < 8; ++i) z[i] = 0.0f;
  return z;
}

__device__ __forceinline__ v16h load_frag(const f16* p, int ld) {
  int lane = threadIdx.x & 31;
  int m = lane & 15, hh = lane >> 4;
  union { v16h v; v8h h[2]; } u;
  u.h[0] = *(const v8h*)(p + m * ld + 8 * hh);
  u.h[1] = *(const v8h*)(p + m * ld + 16 + 8 * hh);
  return u.v;
}

__device__ __forceinline__ v8f wmma16(v16h a, v16h b, v8f c) {
  return __builtin_amdgcn_wmma_f32_16x16x32_f16(false, a, false, b,
                                                (short)0, c, false, false);
}

// ---------------------------------------------------------------------------
// Async global->LDS 64x64 f16 tile staging (blockDim.x == 128).
// Each lane issues 4x GLOBAL_LOAD_ASYNC_TO_LDS_B128 (ASYNCcnt-tracked, no
// VGPR round trip).  LDS operand = low 32 bits of the flat address
// (flat->LDS aperture truncation, ISA 10.2).  Same INST_OFFSET applies to
// both LDS and global sides.
// ---------------------------------------------------------------------------
__device__ __forceinline__ void async_tile64(f16* dst, const f16* src, int ld_src) {
  int t = threadIdx.x;
  int row = t >> 1, c0 = (t & 1) * 32;
  unsigned long long g =
      (unsigned long long)(size_t)(src + (size_t)row * ld_src + c0);
  unsigned l = (unsigned)(size_t)(dst + row * 64 + c0);
  asm volatile(
      "global_load_async_to_lds_b128 %0, %1, off offset:0\n\t"
      "global_load_async_to_lds_b128 %0, %1, off offset:16\n\t"
      "global_load_async_to_lds_b128 %0, %1, off offset:32\n\t"
      "global_load_async_to_lds_b128 %0, %1, off offset:48"
      :: "v"(l), "v"(g) : "memory");
}

__device__ __forceinline__ void wait_async0() {
  asm volatile("s_wait_asynccnt 0x0" ::: "memory");
}

// ---------------------------------------------------------------------------
// Elementwise converters
// ---------------------------------------------------------------------------
__global__ void cvt_f16_kernel(const float* __restrict__ src, f16* __restrict__ dst, int n) {
  int i = blockIdx.x * blockDim.x + threadIdx.x;
  if (i < n) dst[i] = (f16)src[i];
}

// dst[c][r] = src[r][c]  (f32 -> f16 transpose; for weights -> N x K)
__global__ void cvt_f16_T_kernel(const float* __restrict__ src, f16* __restrict__ dst,
                                 int rows, int cols) {
  int i = blockIdx.x * blockDim.x + threadIdx.x;
  if (i >= rows * cols) return;
  int r = i / cols, c = i % cols;
  dst[(size_t)c * rows + r] = (f16)src[(size_t)r * cols + c];
}

// r0 = rel_emb0(64x64) @ Wr0(64x1024) + br0 -> r0h[h][z][d] (f16)
__global__ void r0_kernel(const float* __restrict__ rel_emb0, const float* __restrict__ Wr0,
                          const float* __restrict__ br0, f16* __restrict__ r0h) {
  int i = blockIdx.x * blockDim.x + threadIdx.x;
  if (i >= ZN * ED) return;
  int z = i / ED, e = i % ED;
  float acc = br0[e];
#pragma unroll
  for (int k = 0; k < 64; ++k) acc += rel_emb0[z * 64 + k] * Wr0[k * ED + e];
  int h = e / HD, d = e % HD;
  r0h[((size_t)h * ZN + z) * HD + d] = (f16)acc;
}

// sum_q = emb_sum[tok] * scale -> (B,H,S,D) f16
__global__ void sumq_kernel(const float* __restrict__ emb_sum, const int* __restrict__ tok,
                            f16* __restrict__ out) {
  int i = blockIdx.x * blockDim.x + threadIdx.x;
  if (i >= BS * SL * ED) return;
  int b = i / (SL * ED);
  int rem = i % (SL * ED);
  int s = rem / ED, e = rem % ED;
  int t = tok[b * SL + s];
  float v = emb_sum[(size_t)t * ED + e] * QK_SCALE;
  int h = e / HD, d = e % HD;
  out[(((size_t)(b * NH + h)) * SL + s) * HD + d] = (f16)v;
}

// ---------------------------------------------------------------------------
// Generic WMMA GEMM (async double-buffered LDS pipeline):
//   out = epilogue( A(MxK,f16) @ WT(NxK,f16) + bias )
// modes: 0 = f32 row-major out
//        1 = f16 pack (B,H,R,D) with * QK_SCALE        (Q)
//        2 = per-64 LayerNorm, f16 pack (B,H,R,D)       (K)
//        3 = per-64 LayerNorm, f16 pack (B,H,D,R) (V^T) (V)
// ---------------------------------------------------------------------------
__global__ __launch_bounds__(128)
void gemm_kernel(const f16* __restrict__ A, const f16* __restrict__ WT,
                 const float* __restrict__ bias, void* __restrict__ out,
                 int M, int N, int K, int mode,
                 const float* __restrict__ lng, const float* __restrict__ lnb) {
  __shared__ f16 As[2][64 * 64];
  __shared__ f16 Bs[2][64 * 64];
  int m0 = blockIdx.y * 64, n0 = blockIdx.x * 64;
  int wave = threadIdx.x >> 5, lane = threadIdx.x & 31;

  v8f acc[4];
#pragma unroll
  for (int f = 0; f < 4; ++f) acc[f] = zero_v8f();

  const int nk = K >> 6;
  async_tile64(As[0], A + (size_t)m0 * K, K);
  async_tile64(Bs[0], WT + (size_t)n0 * K, K);

  for (int i = 0; i < nk; ++i) {
    int cur = i & 1;
    wait_async0();       // tile i's async group (own lanes) complete
    __syncthreads();     // everyone's writes landed; buffer cur^1 is free
    if (i + 1 < nk) {
      async_tile64(As[cur ^ 1], A + (size_t)m0 * K + (i + 1) * 64, K);
      async_tile64(Bs[cur ^ 1], WT + (size_t)n0 * K + (i + 1) * 64, K);
    }
    v16h a0 = load_frag(As[cur] + wave * 16 * 64, 64);
    v16h a1 = load_frag(As[cur] + wave * 16 * 64 + 32, 64);
#pragma unroll
    for (int f = 0; f < 4; ++f) {
      v16h b0 = load_frag(Bs[cur] + f * 16 * 64, 64);
      v16h b1 = load_frag(Bs[cur] + f * 16 * 64 + 32, 64);
      acc[f] = wmma16(a0, b0, acc[f]);
      acc[f] = wmma16(a1, b1, acc[f]);
    }
  }

  int rbase = wave * 16 + 8 * (lane >> 4);
  int ncol = lane & 15;
#pragma unroll
  for (int f = 0; f < 4; ++f)
#pragma unroll
    for (int j = 0; j < 8; ++j) acc[f][j] += bias[n0 + f * 16 + ncol];

  if (mode >= 2) {  // LayerNorm over the 64 cols of this tile (== one head)
    float mean[8], rstd[8];
#pragma unroll
    for (int j = 0; j < 8; ++j) {
      float s = acc[0][j] + acc[1][j] + acc[2][j] + acc[3][j];
#pragma unroll
      for (int msk = 1; msk <= 8; msk <<= 1) s += __shfl_xor(s, msk, 32);
      mean[j] = s * (1.0f / 64.0f);
    }
#pragma unroll
    for (int j = 0; j < 8; ++j) {
      float s = 0.0f;
#pragma unroll
      for (int f = 0; f < 4; ++f) {
        float d = acc[f][j] - mean[j];
        s += d * d;
      }
#pragma unroll
      for (int msk = 1; msk <= 8; msk <<= 1) s += __shfl_xor(s, msk, 32);
      rstd[j] = rsqrtf(s * (1.0f / 64.0f) + LN_EPS);
    }
#pragma unroll
    for (int f = 0; f < 4; ++f) {
      int d = f * 16 + ncol;
#pragma unroll
      for (int j = 0; j < 8; ++j)
        acc[f][j] = (acc[f][j] - mean[j]) * rstd[j] * lng[d] + lnb[d];
    }
  }

  if (mode == 0) {
    float* O = (float*)out;
#pragma unroll
    for (int f = 0; f < 4; ++f)
#pragma unroll
      for (int j = 0; j < 8; ++j)
        O[(size_t)(m0 + rbase + j) * N + n0 + f * 16 + ncol] = acc[f][j];
  } else {
    f16* O = (f16*)out;
    int h = n0 >> 6;
#pragma unroll
    for (int f = 0; f < 4; ++f) {
      int d = f * 16 + ncol;
#pragma unroll
      for (int j = 0; j < 8; ++j) {
        int gm = m0 + rbase + j;
        int r = gm / BS, b = gm % BS;
        float v = acc[f][j];
        if (mode == 1) v *= QK_SCALE;
        if (mode == 3)
          O[(((size_t)(b * NH + h)) * HD + d) * RL + r] = (f16)v;      // V^T
        else
          O[(((size_t)(b * NH + h)) * RL + r) * HD + d] = (f16)v;      // Q / K
      }
    }
  }
}

// ---------------------------------------------------------------------------
// Fused flash attention with RPE bias gather (async double-buffered K/V).
//   sum mode  (has_sum_prefix=0, outf!=0): Q=sum_q (S rows), keys = reg keys
//             with bias (rel row_off=0), out f32 (BH,S,D)
//   reg mode  (has_sum_prefix=1, outh!=0): Q=reg_q (R rows), keys = [sum_k2
//             (no bias)] ++ [reg keys with bias, rel row_off=S],
//             out f16 attn (R*B, E)
// Bias table qr = Q_tile @ r0^T computed once per block into LDS via WMMA.
// ---------------------------------------------------------------------------
struct ChunkSrc { const f16* k; const f16* v; int ldv; int kb; int bias; };

__device__ __forceinline__ ChunkSrc chunk_src(int c, int has_prefix, int bh,
                                              const f16* kreg, const f16* vTreg,
                                              const f16* ksum, const f16* vTsum) {
  ChunkSrc r;
  if (has_prefix && c == 0) {
    r.k = ksum + (size_t)bh * SL * HD;
    r.v = vTsum + (size_t)bh * HD * SL;
    r.ldv = SL; r.bias = 0; r.kb = 0;
  } else {
    int k0 = (c - has_prefix) * 64;
    r.k = kreg + ((size_t)bh * RL + k0) * HD;
    r.v = vTreg + (size_t)bh * HD * RL + k0;
    r.ldv = RL; r.bias = 1; r.kb = k0;
  }
  return r;
}

__global__ __launch_bounds__(128)
void attn_kernel(const f16* __restrict__ qsrc,   // (BH, Lq, 64)
                 const f16* __restrict__ kreg,   // (BH, R, 64)
                 const f16* __restrict__ vTreg,  // (BH, 64, R)
                 const f16* __restrict__ ksum,   // (BH, S, 64) or null
                 const f16* __restrict__ vTsum,  // (BH, 64, S) or null
                 const f16* __restrict__ r0h,    // (H, Z, 64)
                 const int* __restrict__ rel,    // (B, S+R, R)
                 float* __restrict__ outf, f16* __restrict__ outh,
                 int Lq, int row_off, int has_sum_prefix) {
  __shared__ f16 Qs[64 * 64];
  __shared__ f16 Ks[2][64 * 64];
  __shared__ f16 Vs[2][64 * 64];
  __shared__ float QR[64 * 64];
  __shared__ f16 Ps[4][16 * 64];

  int bh = blockIdx.y;
  int b = bh / NH, h = bh % NH;
  int q0 = blockIdx.x * 64;
  int wave = threadIdx.x >> 5, lane = threadIdx.x & 31;
  int nchunks = (has_sum_prefix ? 1 : 0) + RL / 64;

  // Prologue: stage Q, the r0 head tile, and chunk 0's K/V in one async group.
  async_tile64(Qs, qsrc + ((size_t)bh * Lq + q0) * HD, HD);
  async_tile64(Ks[0], r0h + (size_t)h * ZN * HD, HD);  // r0 tile [z][d]
  {
    ChunkSrc c0 = chunk_src(0, has_sum_prefix, bh, kreg, vTreg, ksum, vTsum);
    async_tile64(Ks[1], c0.k, HD);     // chunk c lives in buffer (c+1)&1
    async_tile64(Vs[1], c0.v, c0.ldv);
  }
  wait_async0();
  __syncthreads();

  v16h qa0 = load_frag(Qs + wave * 16 * 64, 64);
  v16h qa1 = load_frag(Qs + wave * 16 * 64 + 32, 64);

  int rbase = wave * 16 + 8 * (lane >> 4);
  int ncol = lane & 15;

  {  // qr = Q @ r0^T  (64 x 64) into LDS
#pragma unroll
    for (int f = 0; f < 4; ++f) {
      v8f c = zero_v8f();
      v16h b0 = load_frag(Ks[0] + f * 16 * 64, 64);
      v16h b1 = load_frag(Ks[0] + f * 16 * 64 + 32, 64);
      c = wmma16(qa0, b0, c);
      c = wmma16(qa1, b1, c);
#pragma unroll
      for (int j = 0; j < 8; ++j)
        QR[(rbase + j) * 64 + f * 16 + ncol] = c[j];
    }
  }

  float ms[8], ls[8];
#pragma unroll
  for (int j = 0; j < 8; ++j) { ms[j] = -1e30f; ls[j] = 0.0f; }
  v8f o[4];
#pragma unroll
  for (int f = 0; f < 4; ++f) o[f] = zero_v8f();

  for (int c = 0; c < nchunks; ++c) {
    ChunkSrc cs = chunk_src(c, has_sum_prefix, bh, kreg, vTreg, ksum, vTsum);
    int cur = (c + 1) & 1;  // buffer holding chunk c

    wait_async0();     // chunk c's async group (own lanes) complete
    __syncthreads();   // all writes landed; buffer cur^1 free (and QR done, c==0)
    if (c + 1 < nchunks) {
      ChunkSrc ns = chunk_src(c + 1, has_sum_prefix, bh, kreg, vTreg, ksum, vTsum);
      async_tile64(Ks[cur ^ 1], ns.k, HD);
      async_tile64(Vs[cur ^ 1], ns.v, ns.ldv);
    }

    // scores S = Q @ K^T (Ks stored [key][d])
    v8f s4[4];
#pragma unroll
    for (int f = 0; f < 4; ++f) {
      v16h b0 = load_frag(Ks[cur] + f * 16 * 64, 64);
      v16h b1 = load_frag(Ks[cur] + f * 16 * 64 + 32, 64);
      v8f s = zero_v8f();
      s = wmma16(qa0, b0, s);
      s = wmma16(qa1, b1, s);
      s4[f] = s;
    }

    if (cs.bias) {
#pragma unroll
      for (int f = 0; f < 4; ++f) {
        int kg = cs.kb + f * 16 + ncol;
        const int* relrow =
            rel + ((size_t)b * (SL + RL) + row_off + q0) * RL + kg;
#pragma unroll
        for (int j = 0; j < 8; ++j) {
          int row = rbase + j;
          int idx = relrow[(size_t)row * RL];
          s4[f][j] += QR[row * 64 + idx];
        }
      }
    }

    // online softmax (row = 16-lane half of the wave)
    float pscale[8];
#pragma unroll
    for (int j = 0; j < 8; ++j) {
      float v = fmaxf(fmaxf(s4[0][j], s4[1][j]), fmaxf(s4[2][j], s4[3][j]));
#pragma unroll
      for (int msk = 1; msk <= 8; msk <<= 1) v = fmaxf(v, __shfl_xor(v, msk, 32));
      float mnew = fmaxf(ms[j], v);
      pscale[j] = __expf(ms[j] - mnew);
      ms[j] = mnew;
    }
#pragma unroll
    for (int f = 0; f < 4; ++f)
#pragma unroll
      for (int j = 0; j < 8; ++j) s4[f][j] = __expf(s4[f][j] - ms[j]);
#pragma unroll
    for (int j = 0; j < 8; ++j) {
      float v = s4[0][j] + s4[1][j] + s4[2][j] + s4[3][j];
#pragma unroll
      for (int msk = 1; msk <= 8; msk <<= 1) v += __shfl_xor(v, msk, 32);
      ls[j] = ls[j] * pscale[j] + v;
    }
#pragma unroll
    for (int f = 0; f < 4; ++f)
#pragma unroll
      for (int j = 0; j < 8; ++j) o[f][j] *= pscale[j];

    // P -> LDS (per-wave) to re-layout as A fragment, then O += P @ V
    f16* pw = Ps[wave];
#pragma unroll
    for (int f = 0; f < 4; ++f)
#pragma unroll
      for (int j = 0; j < 8; ++j)
        pw[(j + 8 * (lane >> 4)) * 64 + f * 16 + ncol] = (f16)s4[f][j];
    v16h pa0 = load_frag(pw, 64);
    v16h pa1 = load_frag(pw + 32, 64);
#pragma unroll
    for (int f = 0; f < 4; ++f) {
      v16h b0 = load_frag(Vs[cur] + f * 16 * 64, 64);   // Vs stored [d][key]
      v16h b1 = load_frag(Vs[cur] + f * 16 * 64 + 32, 64);
      o[f] = wmma16(pa0, b0, o[f]);
      o[f] = wmma16(pa1, b1, o[f]);
    }
  }

  float inv[8];
#pragma unroll
  for (int j = 0; j < 8; ++j) inv[j] = 1.0f / ls[j];

  if (outf) {
#pragma unroll
    for (int f = 0; f < 4; ++f)
#pragma unroll
      for (int j = 0; j < 8; ++j) {
        int row = q0 + rbase + j;
        outf[((size_t)bh * Lq + row) * HD + f * 16 + ncol] = o[f][j] * inv[j];
      }
  } else {
#pragma unroll
    for (int f = 0; f < 4; ++f) {
      int d = f * 16 + ncol;
#pragma unroll
      for (int j = 0; j < 8; ++j) {
        int r = q0 + rbase + j;
        outh[((size_t)r * BS + b) * ED + h * HD + d] = (f16)(o[f][j] * inv[j]);
      }
    }
  }
}

// ---------------------------------------------------------------------------
// sum_k2 / sum_v2: one wave per (bh,s) row: y = LN(x @ W2 + b2); f16 pack.
// transposed=0 -> (BH,S,64); transposed=1 -> (BH,64,S)
// ---------------------------------------------------------------------------
__global__ __launch_bounds__(128)
void k2v2_kernel(const float* __restrict__ x, const float* __restrict__ W2,
                 const float* __restrict__ b2, const float* __restrict__ g,
                 const float* __restrict__ bb, f16* __restrict__ out,
                 int transposed) {
  int wid = blockIdx.x * 4 + (threadIdx.x >> 5);
  int lane = threadIdx.x & 31;
  if (wid >= BS * NH * SL) return;
  const float* xr = x + (size_t)wid * 64;
  float x0 = xr[lane], x1 = xr[lane + 32];
  float acc0 = b2[lane], acc1 = b2[lane + 32];
#pragma unroll
  for (int k = 0; k < 64; ++k) {
    float xv = (k < 32) ? __shfl(x0, k, 32) : __shfl(x1, k - 32, 32);
    acc0 += xv * W2[k * 64 + lane];
    acc1 += xv * W2[k * 64 + lane + 32];
  }
  float s = acc0 + acc1;
#pragma unroll
  for (int msk = 1; msk <= 16; msk <<= 1) s += __shfl_xor(s, msk, 32);
  float mean = s * (1.0f / 64.0f);
  float d0 = acc0 - mean, d1 = acc1 - mean;
  float vv = d0 * d0 + d1 * d1;
#pragma unroll
  for (int msk = 1; msk <= 16; msk <<= 1) vv += __shfl_xor(vv, msk, 32);
  float rstd = rsqrtf(vv * (1.0f / 64.0f) + LN_EPS);
  float y0 = d0 * rstd * g[lane] + bb[lane];
  float y1 = d1 * rstd * g[lane + 32] + bb[lane + 32];
  int bh = wid / SL, si = wid % SL;
  if (!transposed) {
    out[((size_t)bh * SL + si) * 64 + lane] = (f16)y0;
    out[((size_t)bh * SL + si) * 64 + lane + 32] = (f16)y1;
  } else {
    out[((size_t)bh * 64 + lane) * SL + si] = (f16)y0;
    out[((size_t)bh * 64 + lane + 32) * SL + si] = (f16)y1;
  }
}

// ---------------------------------------------------------------------------
extern "C" void kernel_launch(void* const* d_in, const int* in_sizes, int n_in,
                              void* d_out, int out_size, void* d_ws, size_t ws_size,
                              hipStream_t stream) {
  const float* reg_x   = (const float*)d_in[0];
  const int*   tok     = (const int*)d_in[1];
  const int*   rel     = (const int*)d_in[2];
  const float* emb_sum = (const float*)d_in[5];
  const float* rel0    = (const float*)d_in[6];
  const float* Wq = (const float*)d_in[7],  *bq = (const float*)d_in[8];
  const float* Wk = (const float*)d_in[9],  *bk = (const float*)d_in[10];
  const float* Wv = (const float*)d_in[11], *bv = (const float*)d_in[12];
  const float* Wr0 = (const float*)d_in[13], *br0 = (const float*)d_in[14];
  const float* Wk2 = (const float*)d_in[15], *bk2 = (const float*)d_in[16];
  const float* Wv2 = (const float*)d_in[17], *bv2 = (const float*)d_in[18];
  const float* Wo = (const float*)d_in[19], *bo = (const float*)d_in[20];
  const float* lnkg = (const float*)d_in[21], *lnkb = (const float*)d_in[22];
  const float* lnvg = (const float*)d_in[23], *lnvb = (const float*)d_in[24];
  const float* lnk2g = (const float*)d_in[25], *lnk2b = (const float*)d_in[26];
  const float* lnv2g = (const float*)d_in[27], *lnv2b = (const float*)d_in[28];

  size_t off = 0;
  auto carve = [&](size_t bytes) {
    void* p = (char*)d_ws + off;
    off += (bytes + 255) & ~(size_t)255;
    return p;
  };
  const size_t MR = (size_t)RL * BS;            // 4096 rows
  f16* regx_h = (f16*)carve(MR * ED * 2);
  f16* WqT = (f16*)carve((size_t)ED * ED * 2);
  f16* WkT = (f16*)carve((size_t)ED * ED * 2);
  f16* WvT = (f16*)carve((size_t)ED * ED * 2);
  f16* WoT = (f16*)carve((size_t)ED * ED * 2);
  f16* q_h  = (f16*)carve(MR * ED * 2);         // (B,H,R,D)
  f16* k_h  = (f16*)carve(MR * ED * 2);         // (B,H,R,D)
  f16* vT_h = (f16*)carve(MR * ED * 2);         // (B,H,D,R)
  f16* r0_h = (f16*)carve((size_t)NH * ZN * HD * 2);
  f16* sumq_h = (f16*)carve((size_t)BS * NH * SL * HD * 2);
  float* sumx2 = (float*)carve((size_t)BS * NH * SL * HD * 4);
  f16* k2_h  = (f16*)carve((size_t)BS * NH * SL * HD * 2);
  f16* v2T_h = (f16*)carve((size_t)BS * NH * SL * HD * 2);
  f16* attn_h = (f16*)carve(MR * ED * 2);
  (void)ws_size; (void)in_sizes; (void)n_in; (void)out_size;

  int n;
  n = (int)(MR * ED);
  cvt_f16_kernel<<<(n + 255) / 256, 256, 0, stream>>>(reg_x, regx_h, n);
  n = ED * ED;
  cvt_f16_T_kernel<<<(n + 255) / 256, 256, 0, stream>>>(Wq, WqT, ED, ED);
  cvt_f16_T_kernel<<<(n + 255) / 256, 256, 0, stream>>>(Wk, WkT, ED, ED);
  cvt_f16_T_kernel<<<(n + 255) / 256, 256, 0, stream>>>(Wv, WvT, ED, ED);
  cvt_f16_T_kernel<<<(n + 255) / 256, 256, 0, stream>>>(Wo, WoT, ED, ED);
  n = ZN * ED;
  r0_kernel<<<(n + 255) / 256, 256, 0, stream>>>(rel0, Wr0, br0, r0_h);
  n = BS * SL * ED;
  sumq_kernel<<<(n + 255) / 256, 256, 0, stream>>>(emb_sum, tok, sumq_h);

  dim3 ggrid(ED / 64, (int)(MR / 64));
  gemm_kernel<<<ggrid, 128, 0, stream>>>(regx_h, WqT, bq, q_h, (int)MR, ED, ED, 1, nullptr, nullptr);
  gemm_kernel<<<ggrid, 128, 0, stream>>>(regx_h, WkT, bk, k_h, (int)MR, ED, ED, 2, lnkg, lnkb);
  gemm_kernel<<<ggrid, 128, 0, stream>>>(regx_h, WvT, bv, vT_h, (int)MR, ED, ED, 3, lnvg, lnvb);

  // stage 1: summary tokens attend over reg keys (with RPE bias)
  attn_kernel<<<dim3(1, BS * NH), 128, 0, stream>>>(
      sumq_h, k_h, vT_h, nullptr, nullptr, r0_h, rel, sumx2, nullptr,
      SL, /*row_off=*/0, /*has_sum_prefix=*/0);

  k2v2_kernel<<<(BS * NH * SL) / 4, 128, 0, stream>>>(sumx2, Wk2, bk2, lnk2g, lnk2b, k2_h, 0);
  k2v2_kernel<<<(BS * NH * SL) / 4, 128, 0, stream>>>(sumx2, Wv2, bv2, lnv2g, lnv2b, v2T_h, 1);

  // stage 2: reg tokens attend over [sum_k2 ++ reg keys]
  attn_kernel<<<dim3(RL / 64, BS * NH), 128, 0, stream>>>(
      q_h, k_h, vT_h, k2_h, v2T_h, r0_h, rel, nullptr, attn_h,
      RL, /*row_off=*/SL, /*has_sum_prefix=*/1);

  gemm_kernel<<<ggrid, 128, 0, stream>>>(attn_h, WoT, bo, d_out, (int)MR, ED, ED, 0, nullptr, nullptr);
}
